// Word_Attention_15556371546452
// MI455X (gfx1250) — matrix operand order repaired and checked by reading, not compiled, not run on hardware
//
#include <hip/hip_runtime.h>

// ---- CDNA5 wave32 WMMA flash-attention, async-LDS staged K/V tiles ----
// B=4, N=4096, D=256, H=128. fp32 in/out, f16 WMMA operands, f32 accum.

typedef _Float16 v8h  __attribute__((ext_vector_type(8)));
typedef _Float16 v16h __attribute__((ext_vector_type(16)));
typedef float    v8f  __attribute__((ext_vector_type(8)));
typedef float    v4f  __attribute__((ext_vector_type(4)));

#define DEV static __device__ __forceinline__

DEV v16h cat16(v8h lo, v8h hi) {
    return __builtin_shufflevector(lo, hi, 0,1,2,3,4,5,6,7,8,9,10,11,12,13,14,15);
}

DEV v8f wmma_f16(v16h a, v16h b, v8f c) {
    return __builtin_amdgcn_wmma_f32_16x16x32_f16(false, a, false, b, (short)0, c, false, false);
}

// async copy 16B global -> LDS, tracked by ASYNCcnt (gfx1250)
DEV void async_b128(unsigned lds_byte_off, const _Float16* gptr) {
    asm volatile("global_load_async_to_lds_b128 %0, %1, off"
                 :: "v"(lds_byte_off), "v"((unsigned long long)(uintptr_t)gptr)
                 : "memory");
}
DEV unsigned lds_off(const void* p) { return (unsigned)(uintptr_t)p; }

// A-fragment (16 rows x K=32) from row-major f16, row stride ld (halfs).
// lanes 0-15: row lane&15, K {c..c+7, c+16..c+23}; lanes 16-31: K {+8, +24}.
DEV v16h load_a_frag(const _Float16* base, int row0, int ld, int c, int lane) {
    int r  = row0 + (lane & 15);
    int k0 = c + ((lane >> 4) << 3);
    const _Float16* p = base + r * ld + k0;
    return cat16(*(const v8h*)(p), *(const v8h*)(p + 16));
}

// B-fragment (K=32 x 16 cols), col n = source row n. lanes 0-15: K {c..c+7, c+16..c+23}; lanes 16-31: K {+8, +24}.
DEV v16h load_b_frag(const _Float16* base, int col0, int ld, int c, int lane) {
    int n  = col0 + (lane & 15);
    int k0 = c + ((lane >> 4) << 3);
    const _Float16* p = base + n * ld + k0;
    return cat16(*(const v8h*)(p), *(const v8h*)(p + 16));
}

// ---------------- kernel 1: transpose weights to f16 Wt[h][d] ----------------
__global__ __launch_bounds__(256) void prep_w(const float* __restrict__ Wq,
                                              const float* __restrict__ Wk,
                                              const float* __restrict__ Wv,
                                              _Float16* __restrict__ wt) {
    const float* W = (blockIdx.y == 0) ? Wq : (blockIdx.y == 1) ? Wk : Wv;
    _Float16* dst = wt + blockIdx.y * (128 * 256);
    int idx = blockIdx.x * 256 + threadIdx.x;
    int h = idx >> 8;
    int d = idx & 255;
    dst[idx] = (_Float16)W[d * 128 + h];
}

// ---------------- kernel 2: QKV projection via WMMA ----------------
__global__ __launch_bounds__(256) void qkv_proj(const float* __restrict__ x,
                                                const _Float16* __restrict__ wt,
                                                const float* __restrict__ bq,
                                                const float* __restrict__ bk,
                                                const float* __restrict__ bv,
                                                _Float16* __restrict__ q16,
                                                _Float16* __restrict__ k16,
                                                _Float16* __restrict__ vT) {
    const int lane  = threadIdx.x & 31;
    const int wave  = threadIdx.x >> 5;
    const int tile  = blockIdx.x * 8 + wave;
    const int rbase = tile * 16;
    const int hi    = lane >> 4;

    v16h xa[8];
    {
        int r = rbase + (lane & 15);
        const float* xr = x + r * 256;
#pragma unroll
        for (int c = 0; c < 8; ++c) {
            int d0 = c * 32 + hi * 8;
            v4f f0 = *(const v4f*)(xr + d0);
            v4f f1 = *(const v4f*)(xr + d0 + 4);
            v4f f2 = *(const v4f*)(xr + d0 + 16);
            v4f f3 = *(const v4f*)(xr + d0 + 20);
            v16h a;
#pragma unroll
            for (int i = 0; i < 4; ++i) {
                a[i]      = (_Float16)f0[i];
                a[4 + i]  = (_Float16)f1[i];
                a[8 + i]  = (_Float16)f2[i];
                a[12 + i] = (_Float16)f3[i];
            }
            xa[c] = a;
        }
    }

#pragma unroll
    for (int m = 0; m < 3; ++m) {
        const _Float16* w = wt + m * (128 * 256);
        const float* bias = (m == 0) ? bq : (m == 1) ? bk : bv;
#pragma unroll
        for (int ht = 0; ht < 8; ++ht) {
            // batch all 16 loads before the 8-WMMA chain
            v16h bw[8];
#pragma unroll
            for (int c = 0; c < 8; ++c)
                bw[c] = load_b_frag(w, ht * 16, 256, c * 32, lane);
            v8f acc = {};
#pragma unroll
            for (int c = 0; c < 8; ++c) acc = wmma_f16(xa[c], bw[c], acc);

            int h = ht * 16 + (lane & 15);
            float bb = bias[h];
#pragma unroll
            for (int i = 0; i < 8; ++i) acc[i] += bb;

            if (m == 2) {
                v8h pk;
#pragma unroll
                for (int i = 0; i < 8; ++i) pk[i] = (_Float16)acc[i];
                *(v8h*)(vT + h * 16384 + rbase + hi * 8) = pk;
            } else {
                _Float16* dst = (m == 0) ? q16 : k16;
#pragma unroll
                for (int i = 0; i < 8; ++i)
                    dst[(rbase + i + 8 * hi) * 128 + h] = (_Float16)acc[i];
            }
        }
    }
}

// ---------------- kernel 3: flash attention ----------------
// Block = 8 waves = 8 query tiles of the same batch. K/V 32-key tiles staged
// into double-buffered LDS with global_load_async_to_lds_b128 (ASYNCcnt).
__global__ __launch_bounds__(256) void flash_attn(const _Float16* __restrict__ q16,
                                                  const _Float16* __restrict__ k16,
                                                  const _Float16* __restrict__ vT,
                                                  float* __restrict__ out) {
    __shared__ _Float16 ksh[2][32][128];   // 2 x 8KB: 32 keys x 128 h
    __shared__ _Float16 vsh[2][128][32];   // 2 x 8KB: 128 h x 32 keys (from vT)

    const int t     = threadIdx.x;
    const int lane  = t & 31;
    const int wave  = t >> 5;
    const int b     = blockIdx.x >> 5;                    // 32 blocks per batch
    const int qb    = ((blockIdx.x & 31) * 8 + wave) * 16;
    const int nbase = b * 4096;
    const int hi    = lane >> 4;

    // issue one 32-key stage (4 async b128 per thread = 16KB per block)
    auto stage = [&](int jj, int buf) {
        const int kb = jj * 32;
        int c = t;                                        // k tile chunk 0..255
        async_b128(lds_off(&ksh[buf][c >> 4][(c & 15) * 8]),
                   k16 + (nbase + kb + (c >> 4)) * 128 + (c & 15) * 8);
        c = t + 256;                                      // k tile chunk 256..511
        async_b128(lds_off(&ksh[buf][c >> 4][(c & 15) * 8]),
                   k16 + (nbase + kb + (c >> 4)) * 128 + (c & 15) * 8);
        c = t;                                            // v tile chunk 0..255
        async_b128(lds_off(&vsh[buf][c >> 2][(c & 3) * 8]),
                   vT + (c >> 2) * 16384 + nbase + kb + (c & 3) * 8);
        c = t + 256;                                      // v tile chunk 256..511
        async_b128(lds_off(&vsh[buf][c >> 2][(c & 3) * 8]),
                   vT + (c >> 2) * 16384 + nbase + kb + (c & 3) * 8);
    };

    // Q as B-fragments, resident in VGPRs for the whole pass
    v16h bqf[4];
#pragma unroll
    for (int c = 0; c < 4; ++c)
        bqf[c] = load_b_frag(q16, nbase + qb, 128, c * 32, lane);

    v8f O[8];
#pragma unroll
    for (int h = 0; h < 8; ++h) O[h] = (v8f){};
    float mrow = -1e30f, lrow = 0.0f;

    stage(0, 0);
    stage(1, 1);

    for (int j = 0; j < 128; ++j) {                       // 32 keys / iteration
        const int buf = j & 1;
        // current stage (4 asyncs older than the in-flight next stage) done:
        if (j >= 126) asm volatile("s_wait_asynccnt 0x0" ::: "memory");
        else          asm volatile("s_wait_asynccnt 0x4" ::: "memory");
        __syncthreads();

        // S^T tiles from LDS: rows = keys, cols = queries
        v16h ka0[4], ka1[4];
#pragma unroll
        for (int c = 0; c < 4; ++c) {
            int k0 = c * 32 + hi * 8;
            {
                const _Float16* p = &ksh[buf][lane & 15][k0];
                ka0[c] = cat16(*(const v8h*)p, *(const v8h*)(p + 16));
            }
            {
                const _Float16* p = &ksh[buf][16 + (lane & 15)][k0];
                ka1[c] = cat16(*(const v8h*)p, *(const v8h*)(p + 16));
            }
        }
        v8f S0 = {}, S1 = {};
#pragma unroll
        for (int c = 0; c < 4; ++c) S0 = wmma_f16(ka0[c], bqf[c], S0);
#pragma unroll
        for (int c = 0; c < 4; ++c) S1 = wmma_f16(ka1[c], bqf[c], S1);

        // online softmax per query column (in-lane + one xor-16 shuffle)
        float vmax = S0[0];
#pragma unroll
        for (int i = 1; i < 8; ++i) vmax = fmaxf(vmax, S0[i]);
#pragma unroll
        for (int i = 0; i < 8; ++i) vmax = fmaxf(vmax, S1[i]);
        vmax = fmaxf(vmax, __shfl_xor(vmax, 16, 32));
        float mnew  = fmaxf(mrow, vmax);
        float scale = __expf(mrow - mnew);

        float p0[8], p1[8], rsum = 0.0f;
#pragma unroll
        for (int i = 0; i < 8; ++i) {
            p0[i] = __expf(S0[i] - mnew);
            p1[i] = __expf(S1[i] - mnew);
            rsum += p0[i] + p1[i];
        }
        rsum += __shfl_xor(rsum, 16, 32);
        lrow = lrow * scale + rsum;
        mrow = mnew;

        // P^T D-layout == PV A-fragment layout: pack in-lane
        v16h pa;
#pragma unroll
        for (int i = 0; i < 8; ++i) {
            pa[i]     = (_Float16)p0[i];
            pa[8 + i] = (_Float16)p1[i];
        }

        // rescale O (scale for query row i+8*hi lives in lane i+8*hi)
#pragma unroll
        for (int i = 0; i < 8; ++i) {
            float sc = __shfl(scale, i + 8 * hi, 32);
#pragma unroll
            for (int ht = 0; ht < 8; ++ht) O[ht][i] *= sc;
        }

        // O += P*V ; batch all 16 ds loads before the 8-WMMA chain
        v16h bv[8];
#pragma unroll
        for (int ht = 0; ht < 8; ++ht) {
            const _Float16* p = &vsh[buf][ht * 16 + (lane & 15)][hi * 8];
            bv[ht] = cat16(*(const v8h*)p, *(const v8h*)(p + 16));
        }
#pragma unroll
        for (int ht = 0; ht < 8; ++ht) O[ht] = wmma_f16(pa, bv[ht], O[ht]);

        __syncthreads();                       // everyone done reading buf
        if (j + 2 < 128) stage(j + 2, buf);    // refill the freed buffer
    }

    // epilogue: divide by row sums, write fp32 out[b][n][h]
#pragma unroll
    for (int i = 0; i < 8; ++i) {
        float li  = __shfl(lrow, i + 8 * hi, 32);
        float inv = 1.0f / li;
        int row = nbase + qb + i + 8 * hi;
#pragma unroll
        for (int ht = 0; ht < 8; ++ht) {
            int h = ht * 16 + (lane & 15);
            out[row * 128 + h] = O[ht][i] * inv;
        }
    }
}

extern "C" void kernel_launch(void* const* d_in, const int* in_sizes, int n_in,
                              void* d_out, int out_size, void* d_ws, size_t ws_size,
                              hipStream_t stream) {
    (void)in_sizes; (void)n_in; (void)out_size; (void)ws_size;
    const float* x  = (const float*)d_in[0];
    const float* Wq = (const float*)d_in[1];
    const float* bq = (const float*)d_in[2];
    const float* Wk = (const float*)d_in[3];
    const float* bk = (const float*)d_in[4];
    const float* Wv = (const float*)d_in[5];
    const float* bv = (const float*)d_in[6];
    float* out = (float*)d_out;

    _Float16* ws  = (_Float16*)d_ws;
    _Float16* wt  = ws;                      // 3 * 128*256 f16
    _Float16* q16 = ws + 3 * 128 * 256;      // [16384][128]
    _Float16* k16 = q16 + 16384 * 128;       // [16384][128]
    _Float16* vT  = k16 + 16384 * 128;       // [128][16384]

    prep_w    <<<dim3(128, 3), 256, 0, stream>>>(Wq, Wk, Wv, wt);
    qkv_proj  <<<128, 256, 0, stream>>>(x, wt, bq, bk, bv, q16, k16, vT);
    flash_attn<<<128, 256, 0, stream>>>(q16, k16, vT, out);
}
